// ODEFunc_88321707475429
// MI455X (gfx1250) — matrix operand order repaired and checked by reading, not compiled, not run on hardware
//
#include <hip/hip_runtime.h>

typedef __attribute__((ext_vector_type(16))) _Float16 v16h;
typedef __attribute__((ext_vector_type(8)))  _Float16 v8h;
typedef __attribute__((ext_vector_type(8)))  float    v8f;
typedef __attribute__((ext_vector_type(4)))  float    v4f;

#define NNODES 8192
#define BATCH  32
#define LDIM   64
#define CB     2048       // LDIM * BATCH
#define BN     262144     // BATCH * NNODES
#define NNZ    262144     // NNODES * 32
#define NMAT   5
#define KDIM   320        // LDIM * NMAT
#define BSTRIDE 328       // padded LDS row stride (halves) -> conflict-free ds_load_b128

// ---------------- CSR build ----------------

__global__ void k_zero_i32(int* __restrict__ p, int n) {
  int i = blockIdx.x * blockDim.x + threadIdx.x;
  if (i < n) p[i] = 0;
}

__global__ void k_hist(const int* __restrict__ rows, int nnz, int* __restrict__ cnt) {
  int e = blockIdx.x * blockDim.x + threadIdx.x;
  if (e < nnz) atomicAdd(&cnt[rows[e]], 1);
}

// one block, 256 threads: exclusive scan of 8192 counts -> row starts
__global__ void k_scan(const int* __restrict__ cnt, int* __restrict__ start) {
  __shared__ int part[256];
  int tid = threadIdx.x;
  int buf[32];
  int s = 0;
  #pragma unroll
  for (int i = 0; i < 32; ++i) { buf[i] = cnt[tid * 32 + i]; s += buf[i]; }
  part[tid] = s;
  __syncthreads();
  if (tid == 0) {
    int acc = 0;
    for (int i = 0; i < 256; ++i) { int t = part[i]; part[i] = acc; acc += t; }
    start[NNODES] = acc;
  }
  __syncthreads();
  int p = part[tid];
  #pragma unroll
  for (int i = 0; i < 32; ++i) { start[tid * 32 + i] = p; p += buf[i]; }
}

__global__ void k_scatter(const int* __restrict__ rows, const int* __restrict__ cols,
                          const float* __restrict__ vals, int nnz,
                          const int* __restrict__ start, int* __restrict__ cursor,
                          int* __restrict__ ecol, float* __restrict__ eval) {
  int e = blockIdx.x * blockDim.x + threadIdx.x;
  if (e >= nnz) return;
  int r = rows[e];
  int p = start[r] + atomicAdd(&cursor[r], 1);
  ecol[p] = cols[e];
  eval[p] = vals[e];
}

// ---------------- [B,N,64] -> [N, 64*B] transpose (LDS tiled) ----------------

__global__ void k_bnc_to_ncb(const float* __restrict__ in, float* __restrict__ out) {
  __shared__ float tile[32][65];
  int n = blockIdx.x;
  for (int i = threadIdx.x; i < CB; i += blockDim.x) {
    int c = i & 63, b = i >> 6;
    tile[b][c] = in[(long)b * (NNODES * LDIM) + (long)n * LDIM + c];
  }
  __syncthreads();
  for (int i = threadIdx.x; i < CB; i += blockDim.x) {
    int b = i & 31, c = i >> 5;
    out[(long)n * CB + c * BATCH + b] = tile[b][c];
  }
}

// ---------------- row-gather SpMM with fused Chebyshev: out = alpha*S*x + beta*xprev ----

__global__ void k_spmm_cheb(const int* __restrict__ start, const int* __restrict__ ecol,
                            const float* __restrict__ eval, const float* __restrict__ x,
                            const float* __restrict__ xprev, float* __restrict__ out,
                            float alpha, float beta) {
  __shared__ int   sc[64];
  __shared__ float sv[64];
  int n   = blockIdx.x;
  int col = threadIdx.x * 8;     // 256 threads * 8 = 2048 columns
  v4f acc0 = {}, acc1 = {};
  int e0 = start[n], e1 = start[n + 1];
  for (int eb = e0; eb < e1; eb += 64) {
    int cnt = min(64, e1 - eb);
    __syncthreads();
    if ((int)threadIdx.x < cnt) {
      sc[threadIdx.x] = ecol[eb + threadIdx.x];
      sv[threadIdx.x] = eval[eb + threadIdx.x];
    }
    __syncthreads();
    for (int i = 0; i < cnt; ++i) {
      int   c = sc[i];
      float v = sv[i];
      if (i + 1 < cnt) __builtin_prefetch(x + (long)sc[i + 1] * CB + col, 0, 0);
      const v4f* xr = (const v4f*)(x + (long)c * CB + col);
      acc0 += xr[0] * v;
      acc1 += xr[1] * v;
    }
  }
  v4f* orow = (v4f*)(out + (long)n * CB + col);
  if (beta != 0.0f) {
    const v4f* pr = (const v4f*)(xprev + (long)n * CB + col);
    orow[0] = acc0 * alpha + pr[0] * beta;
    orow[1] = acc1 * alpha + pr[1] * beta;
  } else {
    orow[0] = acc0 * alpha;
    orow[1] = acc1 * alpha;
  }
}

// ---------------- pack X to f16 [BN, 320] (k = c*5 + m) ----------------

__global__ void k_pack_x(const float* __restrict__ x0, const float* __restrict__ x1,
                         const float* __restrict__ x2, const float* __restrict__ x3,
                         const float* __restrict__ x4, _Float16* __restrict__ Xh) {
  long t = (long)blockIdx.x * blockDim.x + threadIdx.x;   // BN*64 threads
  int  c = (int)(t & 63);
  long row = t >> 6;                                      // row = b*N + n
  int  b = (int)(row >> 13);
  int  n = (int)(row & 8191);
  long src = (long)n * CB + c * BATCH + b;
  _Float16* dst = Xh + row * KDIM + c * NMAT;
  dst[0] = (_Float16)x0[src];
  dst[1] = (_Float16)x1[src];
  dst[2] = (_Float16)x2[src];
  dst[3] = (_Float16)x3[src];
  dst[4] = (_Float16)x4[src];
}

// ---------------- pack W^T to f16 [64, 320] so B-fragments are contiguous in k ----------

__global__ void k_pack_w(const float* __restrict__ W, _Float16* __restrict__ WhT) {
  int t = blockIdx.x * blockDim.x + threadIdx.x;          // KDIM*64 threads
  if (t >= KDIM * 64) return;
  int o = t & 63, k = t >> 6;
  WhT[o * KDIM + k] = (_Float16)W[k * 64 + o];
}

// ---------------- WMMA GEMM: C[BN,64] = Xh[BN,320] * W[320,64] + bias, fused epilogue ----
// One wave computes a full 16x64 row-stripe: A fragments loaded once, reused for 4 tiles.
// W^T staged in LDS via async global->LDS copy (ASYNCcnt path).
// mode 0: sigmoid -> theta ; mode 1: tanh -> cmat ; mode 2: -theta * tanh -> final output

__global__ __launch_bounds__(128) void k_gemm_wmma(const _Float16* __restrict__ Xh,
                                                   const _Float16* __restrict__ WhT,
                                                   const float* __restrict__ bias,
                                                   const float* __restrict__ theta,
                                                   float* __restrict__ out, int mode) {
  __shared__ _Float16 Bs[64 * BSTRIDE];   // 64 rows x 328 halves (padded) = 41 KB

  // Async copy of the 40KB W^T tile into LDS: 2560 chunks of 16B, ASYNCcnt-tracked.
  for (int i = threadIdx.x; i < 2560; i += blockDim.x) {
    int o = i / 40, ck = (i % 40) * 8;                         // 40 x 8-half chunks per row
    unsigned           lda = (unsigned)(size_t)(Bs + o * BSTRIDE + ck);
    unsigned long long ga  = (unsigned long long)(size_t)(WhT + o * KDIM + ck);
    asm volatile("global_load_async_to_lds_b128 %0, %1, off"
                 :: "v"(lda), "v"(ga) : "memory");
  }
  asm volatile("s_wait_asynccnt 0x0" ::: "memory");
  __syncthreads();

  int wave = blockIdx.x * 4 + ((int)threadIdx.x >> 5);  // 16384 row-tile waves
  int lane = threadIdx.x & 31;
  long row0 = (long)wave * 16;
  int half = lane >> 4;                   // half-wave selector
  int mn   = lane & 15;                   // A row / B,C column within tile

  const _Float16* arow = Xh + (row0 + mn) * KDIM;       // A: 16 rows striped on both half-waves

  v8f acc[4] = {v8f{}, v8f{}, v8f{}, v8f{}};
  #pragma unroll
  for (int kk = 0; kk < KDIM; kk += 32) {
    // A layout (matches B): lanes 0-15 hold K=0..15, lanes 16-31 hold K=16..31
    v8h lo = *(const v8h*)(arow + kk + 16 * half);
    v8h hi = *(const v8h*)(arow + kk + 16 * half + 8);
    v16h a;
    #pragma unroll
    for (int i = 0; i < 8; ++i) { a[i] = lo[i]; a[i + 8] = hi[i]; }
    // B layout: lanes 0-15 -> K=0..15, lanes 16-31 -> K=16..31, contiguous from LDS-resident W^T
    const _Float16* bk = Bs + mn * BSTRIDE + 16 * half + kk;
    #pragma unroll
    for (int t = 0; t < 4; ++t) {
      v16h b = *(const v16h*)(bk + t * (16 * BSTRIDE));
      acc[t] = __builtin_amdgcn_wmma_f32_16x16x32_f16(false, a, false, b, (short)0, acc[t],
                                                      false, false);
    }
  }

  #pragma unroll
  for (int t = 0; t < 4; ++t) {
    float bo = bias[t * 16 + mn];
    #pragma unroll
    for (int r = 0; r < 8; ++r) {
      long idx = (row0 + r + 8 * half) * 64 + t * 16 + mn;   // C: VGPR r -> M = r + 8*half
      float v = acc[t][r] + bo;
      float res;
      if (mode == 0)      res = 1.0f / (1.0f + __expf(-v));
      else if (mode == 1) res = tanhf(v);
      else                res = -theta[idx] * tanhf(v);
      out[idx] = res;
    }
  }
}

// ---------------- orchestration ----------------

extern "C" void kernel_launch(void* const* d_in, const int* in_sizes, int n_in,
                              void* d_out, int out_size, void* d_ws, size_t ws_size,
                              hipStream_t stream) {
  const float* y     = (const float*)d_in[0];
  const int*   rows1 = (const int*)d_in[1];
  const int*   cols1 = (const int*)d_in[2];
  const float* vals1 = (const float*)d_in[3];
  const int*   rows2 = (const int*)d_in[4];
  const int*   cols2 = (const int*)d_in[5];
  const float* vals2 = (const float*)d_in[6];
  const float* Wt    = (const float*)d_in[7];
  const float* bt    = (const float*)d_in[8];
  const float* W0    = (const float*)d_in[9];
  const float* b0    = (const float*)d_in[10];
  const float* Wf    = (const float*)d_in[11];
  float* out = (float*)d_out;

  char* ws = (char*)d_ws;
  size_t off = 0;
  auto take = [&](size_t bytes) -> char* {
    char* p = ws + off;
    off = (off + bytes + 255) & ~(size_t)255;
    return p;
  };

  float* xs0 = (float*)take((size_t)NNODES * CB * 4);
  float* xs1 = (float*)take((size_t)NNODES * CB * 4);
  float* xs2 = (float*)take((size_t)NNODES * CB * 4);
  float* xs3 = (float*)take((size_t)NNODES * CB * 4);
  float* xs4 = (float*)take((size_t)NNODES * CB * 4);
  _Float16* Xh  = (_Float16*)take((size_t)BN * KDIM * 2);
  _Float16* WhT = (_Float16*)take((size_t)KDIM * 64 * 2);
  float* theta = (float*)take((size_t)BN * 64 * 4);
  float* cmat  = (float*)take((size_t)BN * 64 * 4);
  int*   start1 = (int*)take((size_t)(NNODES + 1) * 4);
  int*   cnt1   = (int*)take((size_t)NNODES * 4);
  int*   ecol1  = (int*)take((size_t)NNZ * 4);
  float* eval1  = (float*)take((size_t)NNZ * 4);
  int*   start2 = (int*)take((size_t)(NNODES + 1) * 4);
  int*   cnt2   = (int*)take((size_t)NNODES * 4);
  int*   ecol2  = (int*)take((size_t)NNZ * 4);
  float* eval2  = (float*)take((size_t)NNZ * 4);

  dim3 B256(256), B128(128);
  const int GB_EDGE = NNZ / 256;       // 1024
  const int GB_PACK = (BN * 64) / 256; // 65536
  const int GB_GEMM = (BN / 16) / 4;   // 4096 blocks * 4 waves = 16384 row stripes

  // ---- CSR for support1 ----
  k_zero_i32<<<32, B256, 0, stream>>>(cnt1, NNODES);
  k_hist<<<GB_EDGE, B256, 0, stream>>>(rows1, NNZ, cnt1);
  k_scan<<<1, B256, 0, stream>>>(cnt1, start1);
  k_zero_i32<<<32, B256, 0, stream>>>(cnt1, NNODES);
  k_scatter<<<GB_EDGE, B256, 0, stream>>>(rows1, cols1, vals1, NNZ, start1, cnt1, ecol1, eval1);
  // ---- CSR for support2 ----
  k_zero_i32<<<32, B256, 0, stream>>>(cnt2, NNODES);
  k_hist<<<GB_EDGE, B256, 0, stream>>>(rows2, NNZ, cnt2);
  k_scan<<<1, B256, 0, stream>>>(cnt2, start2);
  k_zero_i32<<<32, B256, 0, stream>>>(cnt2, NNODES);
  k_scatter<<<GB_EDGE, B256, 0, stream>>>(rows2, cols2, vals2, NNZ, start2, cnt2, ecol2, eval2);

  // ---- gconv over x: diffusion pyramid ----
  k_bnc_to_ncb<<<NNODES, B256, 0, stream>>>(y, xs0);
  k_spmm_cheb<<<NNODES, B256, 0, stream>>>(start1, ecol1, eval1, xs0, xs0, xs1, 1.0f, 0.0f);
  k_spmm_cheb<<<NNODES, B256, 0, stream>>>(start1, ecol1, eval1, xs1, xs0, xs2, 2.0f, -1.0f);
  k_spmm_cheb<<<NNODES, B256, 0, stream>>>(start2, ecol2, eval2, xs0, xs0, xs3, 1.0f, 0.0f);
  k_spmm_cheb<<<NNODES, B256, 0, stream>>>(start2, ecol2, eval2, xs3, xs0, xs4, 2.0f, -1.0f);
  k_pack_x<<<GB_PACK, B256, 0, stream>>>(xs0, xs1, xs2, xs3, xs4, Xh);

  // theta = sigmoid(X @ Wt + bt)
  k_pack_w<<<80, B256, 0, stream>>>(Wt, WhT);
  k_gemm_wmma<<<GB_GEMM, B128, 0, stream>>>(Xh, WhT, bt, theta, theta, 0);
  // c = tanh(X @ W0 + b0)
  k_pack_w<<<80, B256, 0, stream>>>(W0, WhT);
  k_gemm_wmma<<<GB_GEMM, B128, 0, stream>>>(Xh, WhT, b0, theta, cmat, 1);

  // ---- gconv over c ----
  k_bnc_to_ncb<<<NNODES, B256, 0, stream>>>(cmat, xs0);
  k_spmm_cheb<<<NNODES, B256, 0, stream>>>(start1, ecol1, eval1, xs0, xs0, xs1, 1.0f, 0.0f);
  k_spmm_cheb<<<NNODES, B256, 0, stream>>>(start1, ecol1, eval1, xs1, xs0, xs2, 2.0f, -1.0f);
  k_spmm_cheb<<<NNODES, B256, 0, stream>>>(start2, ecol2, eval2, xs0, xs0, xs3, 1.0f, 0.0f);
  k_spmm_cheb<<<NNODES, B256, 0, stream>>>(start2, ecol2, eval2, xs3, xs0, xs4, 2.0f, -1.0f);
  k_pack_x<<<GB_PACK, B256, 0, stream>>>(xs0, xs1, xs2, xs3, xs4, Xh);

  // out = -theta * tanh(X @ Wf + bt)
  k_pack_w<<<80, B256, 0, stream>>>(Wf, WhT);
  k_gemm_wmma<<<GB_GEMM, B128, 0, stream>>>(Xh, WhT, bt, theta, out, 2);
}
